// IntegralGNN_60559038873704
// MI455X (gfx1250) — matrix operand order repaired and checked by reading, not compiled, not run on hardware
//
#include <hip/hip_runtime.h>

#define N_IN 128
#define HIDF 512   // HEADS*HID = 8*64
#define NEG_SLOPE 0.2f

typedef __attribute__((ext_vector_type(16))) __bf16 v16bf;
typedef __attribute__((ext_vector_type(8)))  __bf16 v8bf;
typedef __attribute__((ext_vector_type(8)))  float  v8f;

typedef unsigned v4u __attribute__((ext_vector_type(4)));
typedef int      v8i __attribute__((ext_vector_type(8)));
typedef int      v4i __attribute__((ext_vector_type(4)));

// ---------------- helpers ----------------

__device__ __forceinline__ __bf16 f2bf(float f) {
  unsigned u = __builtin_bit_cast(unsigned, f);
  u += 0x7FFFu + ((u >> 16) & 1u);                 // round to nearest even
  unsigned short h = (unsigned short)(u >> 16);
  return __builtin_bit_cast(__bf16, h);
}

__device__ __forceinline__ float wred(float v) {
#pragma unroll
  for (int off = 16; off > 0; off >>= 1) v += __shfl_xor(v, off, 32);
  return v;
}

__device__ __forceinline__ float lrelu(float x) { return x > 0.f ? x : NEG_SLOPE * x; }

__device__ __forceinline__ void atomicMaxFloat(float* addr, float val) {
  // sign-split trick: monotone in int-bits for >=0, reversed in uint-bits for <0
  if (val >= 0.f) atomicMax((int*)addr, __float_as_int(val));
  else            atomicMin((unsigned int*)addr, (unsigned int)__float_as_int(val));
}

// ---------------- generic fill ----------------

__global__ void fill_kernel(float* __restrict__ p, float v, long long n) {
  long long i = (long long)blockIdx.x * blockDim.x + threadIdx.x;
  if (i < n) p[i] = v;
}

// ---------------- casts ----------------

__global__ void castW_kernel(const float* __restrict__ w, __bf16* __restrict__ o, long long n) {
  long long i = (long long)blockIdx.x * blockDim.x + threadIdx.x;
  if (i < n) o[i] = f2bf(w[i]);
}

// ---------------- LayerNorm (fp32 in) -> bf16 out ----------------

__global__ void __launch_bounds__(256)
ln_cast_kernel(const float* __restrict__ x, const float* __restrict__ g,
               const float* __restrict__ b, __bf16* __restrict__ out, int n, int K) {
  int w    = (int)((blockIdx.x * blockDim.x + threadIdx.x) >> 5);
  int lane = threadIdx.x & 31;
  if (w >= n) return;
  const float* row = x + (size_t)w * K;
  float s = 0.f, sq = 0.f;
  for (int c = lane; c < K; c += 32) { float v = row[c]; s += v; sq += v * v; }
  s = wred(s); sq = wred(sq);
  float mu   = s / (float)K;
  float var  = sq / (float)K - mu * mu;
  float rinv = rsqrtf(var + 1e-6f);
  __bf16* orow = out + (size_t)w * K;
  for (int c = lane; c < K; c += 32) {
    float v = (row[c] - mu) * rinv * g[c] + b[c];
    orow[c] = f2bf(v);
  }
}

// ---------------- WMMA GEMM: C[n,m] = sum_k A[n,k]*B[m,k]  (A,B bf16; C f32) ----------------
// B panel [NT*16 cols x K] is DMA'd into LDS once per block by the Tensor Data Mover,
// with 4-dword padding per column (TDM pad feature) to dodge LDS bank conflicts.

__device__ __forceinline__ v16bf load_a_frag(const __bf16* p) {
  // CDNA5 16-bit A layout: lane holds K = {kh..kh+7, kh+16..kh+23}
  v8bf lo = *reinterpret_cast<const v8bf*>(p);
  v8bf hi = *reinterpret_cast<const v8bf*>(p + 16);
  v16bf r;
#pragma unroll
  for (int i = 0; i < 8; ++i) { r[i] = lo[i]; r[i + 8] = hi[i]; }
  return r;
}

__device__ __forceinline__ v16bf load_b_frag_lds(const __bf16* p) {
  // CDNA5 16-bit B layout: lane holds K = {kh..kh+7, kh+16..kh+23} of its column
  v8bf lo = *reinterpret_cast<const v8bf*>(p);
  v8bf hi = *reinterpret_cast<const v8bf*>(p + 16);
  v16bf r;
#pragma unroll
  for (int i = 0; i < 8; ++i) { r[i] = lo[i]; r[i + 8] = hi[i]; }
  return r;
}

template <int NT>
__global__ void __launch_bounds__(256)
gemm_bf16_wmma(const __bf16* __restrict__ A, const __bf16* __restrict__ B,
               float* __restrict__ C, int nrows, int K, int ldc) {
  // padded column stride: K bf16 + 8 bf16 (= 4 dwords) of TDM-inserted pad
  __shared__ __bf16 sB[NT * 16 * (512 + 8)];
  const int wave = threadIdx.x >> 5;
  const int lane = threadIdx.x & 31;
  const int n0   = (int)blockIdx.y * (NT * 16);
  const int ldsStride = K + 8;

  if (threadIdx.x == 0) {
    // ---- Tensor DMA descriptor (D#) : 2D tile, dwords as elements ----
    const unsigned Kdw  = (unsigned)K >> 1;        // dwords per column
    const unsigned cols = (unsigned)(NT * 16);
    const unsigned padi = (K == 512) ? 7u : 5u;    // pad every Kdw dwords (log2(Kdw)-1)
    unsigned long long ga =
        (unsigned long long)(size_t)(B + (size_t)n0 * K);          // global byte addr
    unsigned lds_base = (unsigned)(unsigned long long)(size_t)(void*)sB;  // LDS byte addr

    v4u g0;
    g0[0] = 1u;                                                   // count=1 (valid D#)
    g0[1] = lds_base;                                             // lds_addr
    g0[2] = (unsigned)(ga & 0xFFFFFFFFu);                         // global_addr[31:0]
    g0[3] = (unsigned)((ga >> 32) & 0x1FFFFFFu) | (2u << 30);     // global_addr[56:32] | type=2

    v8i g1;
    g1[0] = (int)((2u << 16) | (1u << 20) | (padi << 22) | (3u << 25));
            // data_size=4B | pad_enable | pad_interval | pad_amount=4 dwords
    g1[1] = (int)((Kdw & 0xFFFFu) << 16);                         // tensor_dim0[15:0]
    g1[2] = (int)((Kdw >> 16) | ((cols & 0xFFFFu) << 16));        // tdim0 hi | tensor_dim1 lo
    g1[3] = (int)((cols >> 16) | ((Kdw & 0xFFFFu) << 16));        // tdim1 hi | tile_dim0
    g1[4] = (int)(cols & 0xFFFFu);                                // tile_dim1 (tile_dim2=0)
    g1[5] = (int)Kdw;                                             // tensor_dim0_stride[31:0]
    g1[6] = 0;
    g1[7] = 0;
    v4i g2 = {0, 0, 0, 0};
    v4i g3 = {0, 0, 0, 0};
    v8i g4 = {0, 0, 0, 0, 0, 0, 0, 0};   // extra group (clang-23 6-arg form), unused for 2D
    __builtin_amdgcn_tensor_load_to_lds(g0, g1, g2, g3, g4, 0);
    __builtin_amdgcn_s_wait_tensorcnt(0);
  }
  __syncthreads();   // before any early-out so every resident wave participates

  const int m0 = ((int)blockIdx.x * 8 + wave) * 16;
  if (m0 >= nrows) return;

  const int rsel  = lane & 15;
  const int ahalf = (lane >> 4) << 3;   // A: +8 K for upper half-wave
  const int bhalf = (lane >> 4) << 3;   // B: +8 K for upper half-wave

  const __bf16* arow = A + (size_t)(m0 + rsel) * K + ahalf;

  v8f acc[NT];
#pragma unroll
  for (int t = 0; t < NT; ++t)
#pragma unroll
    for (int i = 0; i < 8; ++i) acc[t][i] = 0.f;

  for (int k0 = 0; k0 < K; k0 += 32) {
    v16bf a = load_a_frag(arow + k0);
#pragma unroll
    for (int t = 0; t < NT; ++t) {
      const __bf16* bcol = sB + (size_t)(t * 16 + rsel) * ldsStride + bhalf + k0;
      v16bf bf = load_b_frag_lds(bcol);
      acc[t] = __builtin_amdgcn_wmma_f32_16x16x32_bf16(
          false, a, false, bf, (short)0, acc[t], false, false);
    }
  }

  const int rofs = (lane >> 4) << 3;   // D rows: +8 for upper half-wave
#pragma unroll
  for (int t = 0; t < NT; ++t) {
    float* cp = C + (size_t)(m0 + rofs) * ldc + n0 + t * 16 + rsel;
#pragma unroll
    for (int r = 0; r < 8; ++r) cp[(size_t)r * ldc] = acc[t][r];
  }
}

// ---------------- attention logits per node ----------------

__global__ void __launch_bounds__(256)
alpha_kernel(const float* __restrict__ hlin, const float* __restrict__ asr,
             const float* __restrict__ adt, float* __restrict__ alsrc,
             float* __restrict__ aldst, int n, int heads, int co, int ldh) {
  int w    = (int)((blockIdx.x * blockDim.x + threadIdx.x) >> 5);
  int lane = threadIdx.x & 31;
  if (w >= n * heads) return;
  int node = w / heads, head = w % heads;
  const float* hp = hlin + (size_t)node * ldh + head * co;
  float s1 = 0.f, s2 = 0.f;
  for (int c = lane; c < co; c += 32) {
    float v = hp[c];
    s1 += v * asr[head * co + c];
    s2 += v * adt[head * co + c];
  }
  s1 = wred(s1); s2 = wred(s2);
  if (lane == 0) { alsrc[w] = s1; aldst[w] = s2; }
}

// ---------------- edge passes: segment max, exp-sum, weighted scatter ----------------

__global__ void edge_max_kernel(const int* __restrict__ ei, int E, int n,
                                const float* __restrict__ alsrc,
                                const float* __restrict__ aldst,
                                float* __restrict__ m, int heads) {
  long long idx = (long long)blockIdx.x * blockDim.x + threadIdx.x;
  long long tot = (long long)(E + n) * heads;
  if (idx >= tot) return;
  int e = (int)(idx / heads), h = (int)(idx % heads);
  int src = (e < E) ? ei[e]     : (e - E);
  int dst = (e < E) ? ei[E + e] : (e - E);
  float v = lrelu(alsrc[(size_t)src * heads + h] + aldst[(size_t)dst * heads + h]);
  atomicMaxFloat(&m[(size_t)dst * heads + h], v);
}

__global__ void edge_sum_kernel(const int* __restrict__ ei, int E, int n,
                                const float* __restrict__ alsrc,
                                const float* __restrict__ aldst,
                                const float* __restrict__ m,
                                float* __restrict__ s, int heads) {
  long long idx = (long long)blockIdx.x * blockDim.x + threadIdx.x;
  long long tot = (long long)(E + n) * heads;
  if (idx >= tot) return;
  int e = (int)(idx / heads), h = (int)(idx % heads);
  int src = (e < E) ? ei[e]     : (e - E);
  int dst = (e < E) ? ei[E + e] : (e - E);
  float v = lrelu(alsrc[(size_t)src * heads + h] + aldst[(size_t)dst * heads + h]);
  float ex = __expf(v - m[(size_t)dst * heads + h]);
  atomicAdd(&s[(size_t)dst * heads + h], ex);
}

__global__ void scatter_kernel(const int* __restrict__ ei, int E, int n,
                               const float* __restrict__ alsrc,
                               const float* __restrict__ aldst,
                               const float* __restrict__ m,
                               const float* __restrict__ s,
                               const float* __restrict__ hlin,
                               float* __restrict__ agg,
                               int heads, int co, int ldh, int ldo) {
  long long idx = (long long)blockIdx.x * blockDim.x + threadIdx.x;
  long long tot = (long long)(E + n) * heads;
  if (idx >= tot) return;
  int e = (int)(idx / heads), h = (int)(idx % heads);
  int src = (e < E) ? ei[e]     : (e - E);
  int dst = (e < E) ? ei[E + e] : (e - E);
  size_t dh = (size_t)dst * heads + h;
  float v = lrelu(alsrc[(size_t)src * heads + h] + aldst[dh]);
  float alpha = __expf(v - m[dh]) / s[dh];
  const float* hp = hlin + (size_t)src * ldh + h * co;
  float* op = agg + (size_t)dst * ldo + h * co;
  for (int c = 0; c < co; ++c) atomicAdd(&op[c], hp[c] * alpha);
}

// ---------------- epilogues ----------------

__global__ void epilogue_mid(const float* __restrict__ agg, const float* __restrict__ bias,
                             const float* __restrict__ xres, float* __restrict__ hout,
                             int n, int F, int addres) {
  long long i = (long long)blockIdx.x * blockDim.x + threadIdx.x;
  if (i >= (long long)n * F) return;
  int c = (int)(i % F);
  float v = agg[i] + bias[c];
  float g = 0.5f * v * (1.f + erff(v * 0.70710678118f));   // exact GELU
  if (addres) g += xres[i];
  hout[i] = g;
}

__global__ void final_kernel(const float* __restrict__ agg, const float* __restrict__ b3,
                             const float* __restrict__ resid, float* __restrict__ out, int n) {
  long long i = (long long)blockIdx.x * blockDim.x + threadIdx.x;
  if (i >= 3LL * n) return;
  int c = (int)(i % 3);
  out[i] = agg[i] + b3[c] + resid[i];
}

// ---------------- input-projection residual: x @ res_W.T + res_b ----------------

__global__ void __launch_bounds__(256)
residual_kernel(const float* __restrict__ x, const float* __restrict__ W,
                const float* __restrict__ b, float* __restrict__ out, int n) {
  int w    = (int)((blockIdx.x * blockDim.x + threadIdx.x) >> 5);
  int lane = threadIdx.x & 31;
  if (w >= n) return;
  const float* row = x + (size_t)w * N_IN;
  float a0 = 0.f, a1 = 0.f, a2 = 0.f;
  for (int c = lane; c < N_IN; c += 32) {
    float v = row[c];
    a0 += v * W[c]; a1 += v * W[N_IN + c]; a2 += v * W[2 * N_IN + c];
  }
  a0 = wred(a0); a1 = wred(a1); a2 = wred(a2);
  if (lane == 0) {
    out[(size_t)w * 3 + 0] = a0 + b[0];
    out[(size_t)w * 3 + 1] = a1 + b[1];
    out[(size_t)w * 3 + 2] = a2 + b[2];
  }
}

// ---------------- host orchestration ----------------

extern "C" void kernel_launch(void* const* d_in, const int* in_sizes, int n_in,
                              void* d_out, int out_size, void* d_ws, size_t ws_size,
                              hipStream_t stream) {
  (void)n_in; (void)out_size; (void)ws_size;
  const int N = in_sizes[0] / N_IN;
  const int E = in_sizes[1] / 2;
  const float* x    = (const float*)d_in[0];
  const int*   ei   = (const int*)d_in[1];
  const float* resW = (const float*)d_in[2];
  const float* resb = (const float*)d_in[3];

  char* wsp = (char*)d_ws;
  size_t off = 0;
  auto alloc = [&](size_t bytes) -> void* {
    void* p = wsp + off; off += (bytes + 255) & ~(size_t)255; return p;
  };
  float*  featA = (float*)alloc((size_t)N * HIDF * 4);
  __bf16* ln16  = (__bf16*)alloc((size_t)N * HIDF * 2);
  float*  hlin  = (float*)alloc((size_t)N * HIDF * 4);
  float*  agg   = (float*)alloc((size_t)N * HIDF * 4);
  float*  alsrc = (float*)alloc((size_t)N * 8 * 4);
  float*  aldst = (float*)alloc((size_t)N * 8 * 4);
  float*  mbuf  = (float*)alloc((size_t)N * 8 * 4);
  float*  sbuf  = (float*)alloc((size_t)N * 8 * 4);
  __bf16* W16   = (__bf16*)alloc((size_t)512 * 512 * 2);
  float*  resid = (float*)alloc((size_t)N * 3 * 4);

  dim3 blk(256);
  auto cdiv = [](long long a, long long b) { return (unsigned)((a + b - 1) / b); };

  residual_kernel<<<cdiv((long long)N * 32, 256), blk, 0, stream>>>(x, resW, resb, resid, N);

  for (int i = 0; i < 4; ++i) {
    const float* lng = (const float*)d_in[4 + 6 * i + 0];
    const float* lnb = (const float*)d_in[4 + 6 * i + 1];
    const float* W   = (const float*)d_in[4 + 6 * i + 2];
    const float* asr = (const float*)d_in[4 + 6 * i + 3];
    const float* adt = (const float*)d_in[4 + 6 * i + 4];
    const float* bia = (const float*)d_in[4 + 6 * i + 5];

    const int Kin   = (i == 0) ? N_IN : HIDF;
    const int heads = (i == 3) ? 1 : 8;
    const int co    = (i == 3) ? 3 : 64;
    const int Wrows = heads * co;              // 512 or 3
    const int ldh   = (i == 3) ? 16 : HIDF;    // hlin leading dim (layer3 padded to 16)
    const int ldo   = (i == 3) ? 3 : HIDF;     // agg leading dim
    const float* hin = (i == 0) ? x : featA;

    // 1) LayerNorm -> bf16
    ln_cast_kernel<<<cdiv((long long)N * 32, 256), blk, 0, stream>>>(hin, lng, lnb, ln16, N, Kin);

    // 2) W -> bf16 (layer3: zero-pad to 16 output rows first)
    if (i == 3)
      fill_kernel<<<cdiv(16 * 512 / 2, 256), blk, 0, stream>>>((float*)W16, 0.f, 16 * 512 / 2);
    castW_kernel<<<cdiv((long long)Wrows * Kin, 256), blk, 0, stream>>>(W, W16, (long long)Wrows * Kin);

    // 3) feature transform via WMMA (B panel staged in LDS by Tensor DMA)
    if (i < 3) {
      dim3 grid(cdiv(N, 128), (unsigned)(Wrows / 64));
      gemm_bf16_wmma<4><<<grid, blk, 0, stream>>>(ln16, W16, hlin, N, Kin, HIDF);
    } else {
      dim3 grid(cdiv(N, 128), 1);
      gemm_bf16_wmma<1><<<grid, blk, 0, stream>>>(ln16, W16, hlin, N, Kin, 16);
    }

    // 4) attention logits
    alpha_kernel<<<cdiv((long long)N * heads * 32, 256), blk, 0, stream>>>(
        hlin, asr, adt, alsrc, aldst, N, heads, co, ldh);

    // 5) segment softmax + aggregate
    fill_kernel<<<cdiv((long long)N * heads, 256), blk, 0, stream>>>(mbuf, -3.402823466e38f, (long long)N * heads);
    fill_kernel<<<cdiv((long long)N * heads, 256), blk, 0, stream>>>(sbuf, 0.f, (long long)N * heads);
    fill_kernel<<<cdiv((long long)N * ldo, 256), blk, 0, stream>>>(agg, 0.f, (long long)N * ldo);

    long long tot = (long long)(E + N) * heads;
    edge_max_kernel<<<cdiv(tot, 256), blk, 0, stream>>>(ei, E, N, alsrc, aldst, mbuf, heads);
    edge_sum_kernel<<<cdiv(tot, 256), blk, 0, stream>>>(ei, E, N, alsrc, aldst, mbuf, sbuf, heads);
    scatter_kernel<<<cdiv(tot, 256), blk, 0, stream>>>(ei, E, N, alsrc, aldst, mbuf, sbuf,
                                                       hlin, agg, heads, co, ldh, ldo);

    // 6) epilogue
    if (i < 3) {
      int addres = (Kin == HIDF) ? 1 : 0;   // residual only when shapes match (layers 1,2)
      epilogue_mid<<<cdiv((long long)N * HIDF, 256), blk, 0, stream>>>(
          agg, bia, featA, featA, N, HIDF, addres);
    } else {
      final_kernel<<<cdiv((long long)N * 3, 256), blk, 0, stream>>>(agg, bia, resid, (float*)d_out, N);
    }
  }
}
